// MoDLayer_34961033789802
// MI455X (gfx1250) — matrix operand order repaired and checked by reading, not compiled, not run on hardware
//
#include <hip/hip_runtime.h>

typedef _Float16 f16;
typedef __attribute__((ext_vector_type(16))) _Float16 v16h;
typedef __attribute__((ext_vector_type(8)))  _Float16 v8h;
typedef __attribute__((ext_vector_type(2)))  _Float16 v2hh;
typedef __attribute__((ext_vector_type(8)))  float    v8f;
typedef __attribute__((ext_vector_type(4)))  int      v4i;

#if defined(__gfx1250__) && __has_builtin(__builtin_amdgcn_global_load_async_to_lds_b128) && \
    __has_builtin(__builtin_amdgcn_s_wait_asynccnt)
#define HAVE_ASYNC 1
#else
#define HAVE_ASYNC 0
#endif

// ---------------------------------------------------------------------------
// WMMA fragment helpers (V_WMMA_F32_16X16X32_F16, wave32)
// A fragment (16x32 f16), lane L: row = L&15, g = L>>4:
//   elems e=0..15 -> K = kbase + 16*g + e     (contiguous 16)
// B fragment (32x16 f16), lane L: col = col0 + (L&15), elems e -> K = kbase + 16*g + e
// C/D: VGPR r, lane L: row = r + 8*(L>>4), col = L&15
// ---------------------------------------------------------------------------
__device__ __forceinline__ v16h load_a_frag(const f16* A, int lda, int row0, int kbase, int lane) {
  int r = row0 + (lane & 15);
  int g = lane >> 4;
  const f16* p0 = A + (long)r * lda + kbase + 16 * g;
  v8h lo = *(const v8h*)p0;
  v8h hi = *(const v8h*)(p0 + 8);
  v16h out;
#pragma unroll
  for (int e = 0; e < 8; ++e) { out[e] = lo[e]; out[e + 8] = hi[e]; }
  return out;
}

// B fragment from row-major [K,N] (strided; used only in attention)
__device__ __forceinline__ v16h load_b_frag(const f16* Bm, int ldb, int kbase, int col0, int lane) {
  int n = col0 + (lane & 15);
  int g = lane >> 4;
  const f16* p = Bm + (long)(kbase + 16 * g) * ldb + n;
  v16h out;
#pragma unroll
  for (int e = 0; e < 16; ++e) out[e] = p[(long)e * ldb];
  return out;
}

// B fragment from LDS stored transposed: BT[col][k], ld = ldt halfs (ldt multiple of 8)
__device__ __forceinline__ v16h load_bT_frag(const f16* BT, int ldt, int col0, int lane) {
  int n = col0 + (lane & 15);
  int g = lane >> 4;
  const f16* p = BT + (long)n * ldt + 16 * g;
  v8h lo = *(const v8h*)p;
  v8h hi = *(const v8h*)(p + 8);
  v16h out;
#pragma unroll
  for (int e = 0; e < 8; ++e) { out[e] = lo[e]; out[e + 8] = hi[e]; }
  return out;
}

__device__ __forceinline__ v8f wmma_f16(v16h a, v16h b, v8f c) {
  return __builtin_amdgcn_wmma_f32_16x16x32_f16(false, a, false, b, (short)0, c, false, false);
}

__device__ __forceinline__ float wave_sum(float v) {
#pragma unroll
  for (int off = 16; off; off >>= 1) v += __shfl_down(v, off, 32);
  return v;
}

__device__ __forceinline__ float sigmoidf_(float x) { return 1.f / (1.f + expf(-x)); }
__device__ __forceinline__ float gelu_tanh(float x) {
  float x3 = x * x * x;
  return 0.5f * x * (1.f + tanhf(0.7978845608028654f * (x + 0.044715f * x3)));
}
__device__ __forceinline__ float bce_logit(float x, float t) {
  return fmaxf(x, 0.f) - x * t + log1pf(expf(-fabsf(x)));
}

// ---------------------------------------------------------------------------
// LDS-tiled WMMA GEMM: C[M,N] = A[M,K](f16 rm) @ W[K,N](f16 rm)
// Block: 256 thr = 8 waves; block tile 64x128, BK=32.
// Wave (wr,wc) in 2x4 grid computes a 32x32 tile = 2x2 WMMA tiles.
// A staged via async global->LDS copy (ASYNCcnt path); B staged TRANSPOSED so
// B-fragment reads are contiguous ds_load_b128.  BT pitch 40 halfs (bank pad +
// keeps 16B alignment for fragment reads).
// ---------------------------------------------------------------------------
#define BT_LD 40
__global__ void __launch_bounds__(256) gemm_lds(const f16* __restrict__ A,
                                                const f16* __restrict__ W,
                                                float* __restrict__ Cf,
                                                f16* __restrict__ Ch,
                                                int M, int N, int K) {
  __shared__ __align__(16) f16 Asm[64 * 32];
  __shared__ __align__(16) f16 BsmT[128 * BT_LD];
  int tid = threadIdx.x;
  int lane = tid & 31, wave = tid >> 5;
  int nbn = N >> 7;
  int bm = blockIdx.x / nbn, bn = blockIdx.x % nbn;
  int row0 = bm << 6, col0 = bn << 7;
  int wr = wave >> 2, wc = wave & 3;

  // per-thread staging coordinates
  int arow = tid >> 2, acol = (tid & 3) << 3;          // A: 64 rows x 32 halfs
  int kr0 = (tid >> 4) << 1;                            // B: k rows kr0, kr0+1 (0..31)
  int bcol = (tid & 15) << 3;                           // B: 8 cols

  v8f acc[2][2] = {};
  for (int kb = 0; kb < K; kb += 32) {
    // ---- A tile -> LDS ----
    {
      const f16* src = A + (long)(row0 + arow) * K + kb + acol;
      f16* dst = &Asm[arow * 32 + acol];
#if HAVE_ASYNC
      __builtin_amdgcn_global_load_async_to_lds_b128(
          (__attribute__((address_space(1))) v4i*)(void*)src,
          (__attribute__((address_space(3))) v4i*)(void*)dst, 0, 0);
#else
      *(v8h*)dst = *(const v8h*)src;
#endif
    }
    // ---- B tile -> LDS (transposed) ----
    {
      const f16* src0 = W + (long)(kb + kr0) * N + col0 + bcol;
      v8h r0 = *(const v8h*)src0;
      v8h r1 = *(const v8h*)(src0 + N);
#pragma unroll
      for (int j = 0; j < 8; ++j) {
        v2hh pr; pr[0] = r0[j]; pr[1] = r1[j];
        *(v2hh*)&BsmT[(bcol + j) * BT_LD + kr0] = pr;   // ds_store_b32
      }
    }
    if (kb + 32 < K) {
      __builtin_prefetch(A + (long)(row0 + arow) * K + kb + 32 + acol, 0, 0);
      __builtin_prefetch(W + (long)(kb + 32 + kr0) * N + col0 + bcol, 0, 0);
    }
#if HAVE_ASYNC
    __builtin_amdgcn_s_wait_asynccnt(0);
#endif
    __syncthreads();

#pragma unroll
    for (int ms = 0; ms < 2; ++ms) {
      v16h af = load_a_frag(Asm, 32, (wr << 5) + (ms << 4), 0, lane);
#pragma unroll
      for (int ns = 0; ns < 2; ++ns) {
        v16h bf = load_bT_frag(BsmT, BT_LD, (wc << 5) + (ns << 4), lane);
        acc[ms][ns] = wmma_f16(af, bf, acc[ms][ns]);
      }
    }
    __syncthreads();
  }

#pragma unroll
  for (int ms = 0; ms < 2; ++ms) {
#pragma unroll
    for (int ns = 0; ns < 2; ++ns) {
      int r0g = row0 + (wr << 5) + (ms << 4) + ((lane >> 4) << 3);
      int cg  = col0 + (wc << 5) + (ns << 4) + (lane & 15);
      if (Cf) {
#pragma unroll
        for (int r = 0; r < 8; ++r) Cf[(long)(r0g + r) * N + cg] = acc[ms][ns][r];
      }
      if (Ch) {
#pragma unroll
        for (int r = 0; r < 8; ++r) Ch[(long)(r0g + r) * N + cg] = (f16)acc[ms][ns][r];
      }
    }
  }
}

// ---------------------------------------------------------------------------
__global__ void f32_to_f16(const float* __restrict__ in, f16* __restrict__ out, int n) {
  int i = blockIdx.x * blockDim.x + threadIdx.x;
  if (i < n) out[i] = (f16)in[i];
}

__global__ void init_accum(float* accum) {
  if (threadIdx.x < 2) accum[threadIdx.x] = 0.f;
}

// scores[b,t] = hidden[b,t,:] . router_w   (one wave per token)
__global__ void router_scores(const float* __restrict__ hs, const float* __restrict__ rw,
                              float* __restrict__ scores, int BT, int D) {
  int wid = (blockIdx.x * blockDim.x + threadIdx.x) >> 5;
  int lane = threadIdx.x & 31;
  if (wid >= BT) return;
  const float* row = hs + (long)wid * D;
  float s = 0.f;
  for (int i = lane; i < D; i += 32) s += row[i] * rw[i];
  s = wave_sum(s);
  if (lane == 0) scores[wid] = s;
}

// Per batch: bitonic sort (desc) of 4096 scores, take top 512, sort indices asc.
__global__ void __launch_bounds__(1024) topk_kernel(const float* __restrict__ scores,
                                                    int* __restrict__ idx_out,
                                                    float* __restrict__ gate,
                                                    int* __restrict__ selpos,
                                                    float* __restrict__ targets,
                                                    int T, int K) {
  int b = blockIdx.x;
  const float* sc = scores + (long)b * T;
  __shared__ float sv[4096];
  __shared__ int   si[4096];
  __shared__ int   top[512];
  int tid = threadIdx.x, nt = blockDim.x;
  for (int i = tid; i < T; i += nt) {
    sv[i] = sc[i]; si[i] = i;
    selpos[(long)b * T + i] = -1;
    targets[(long)b * T + i] = 0.f;
  }
  __syncthreads();
  for (int size = 2; size <= T; size <<= 1) {
    for (int stride = size >> 1; stride > 0; stride >>= 1) {
      for (int t = tid; t < (T >> 1); t += nt) {
        int lo = 2 * stride * (t / stride) + (t % stride);
        int hi = lo + stride;
        bool asc = ((lo & size) != 0);   // overall descending
        float a = sv[lo], c = sv[hi];
        bool sw = asc ? (a > c) : (a < c);
        if (sw) { sv[lo] = c; sv[hi] = a; int ti = si[lo]; si[lo] = si[hi]; si[hi] = ti; }
      }
      __syncthreads();
    }
  }
  if (tid < K) top[tid] = si[tid];
  __syncthreads();
  for (int size = 2; size <= K; size <<= 1) {
    for (int stride = size >> 1; stride > 0; stride >>= 1) {
      for (int t = tid; t < (K >> 1); t += nt) {
        int lo = 2 * stride * (t / stride) + (t % stride);
        int hi = lo + stride;
        bool asc = ((lo & size) == 0);   // ascending
        int a = top[lo], c = top[hi];
        if (asc ? (a > c) : (a < c)) { top[lo] = c; top[hi] = a; }
      }
      __syncthreads();
    }
  }
  if (tid < K) {
    int tv = top[tid];
    idx_out[(long)b * K + tid] = tv;
    gate[(long)b * K + tid] = sigmoidf_(sc[tv]);
    selpos[(long)b * T + tv] = tid;
    targets[(long)b * T + tv] = 1.f;
  }
}

// Predictor head + both BCE losses. One wave per token.
__global__ void loss_kernel(const f16* __restrict__ p1, const float* __restrict__ fc1b,
                            const float* __restrict__ fc2w, const float* __restrict__ fc2b,
                            const float* __restrict__ scores, const float* __restrict__ targets,
                            float* __restrict__ accum, int BT, int NP) {
  int wid = (blockIdx.x * blockDim.x + threadIdx.x) >> 5;
  int lane = threadIdx.x & 31;
  if (wid >= BT) return;
  const f16* row = p1 + (long)wid * NP;
  float dot = 0.f;
  for (int i = lane; i < NP; i += 32) {
    float x = (float)row[i] + fc1b[i];
    dot += gelu_tanh(x) * fc2w[i];
  }
  dot = wave_sum(dot);
  if (lane == 0) {
    float t = targets[wid];
    float logit = dot + fc2b[0];
    atomicAdd(&accum[0], bce_logit(scores[wid], t));
    atomicAdd(&accum[1], bce_logit(logit, t));
  }
}

// Gather selected rows + RMSNorm (block per selected row, 256 thr, D=1024)
__global__ void __launch_bounds__(256) gather_rms(const float* __restrict__ hs,
                                                  const int* __restrict__ idx,
                                                  const float* __restrict__ g1,
                                                  float* __restrict__ sel,
                                                  f16* __restrict__ a16,
                                                  int T, int D, int Ksel) {
  int rowid = blockIdx.x;
  int b = rowid / Ksel;
  int t = idx[rowid];
  const float* src = hs + ((long)b * T + t) * D;
  float* dsel = sel + (long)rowid * D;
  f16* da = a16 + (long)rowid * D;
  int tid = threadIdx.x;
  float v[4]; float ss = 0.f;
#pragma unroll
  for (int c = 0; c < 4; ++c) { int i = tid + c * 256; float x = src[i]; v[c] = x; dsel[i] = x; ss += x * x; }
  __shared__ float red[8];
  __shared__ float stot;
  ss = wave_sum(ss);
  if ((tid & 31) == 0) red[tid >> 5] = ss;
  __syncthreads();
  if (tid == 0) { float s = 0.f; for (int i = 0; i < 8; ++i) s += red[i]; stot = s; }
  __syncthreads();
  float inv = rsqrtf(stot / (float)D + 1e-6f);
#pragma unroll
  for (int c = 0; c < 4; ++c) { int i = tid + c * 256; da[i] = (f16)(v[c] * inv * g1[i]); }
}

// RoPE on q,k + layout change: q16/v16 -> [B,H,S,HD], kT16 -> [B,H,HD,S]
__global__ void rope_kernel(const f16* __restrict__ qp, const f16* __restrict__ kp,
                            const f16* __restrict__ vp, const int* __restrict__ idx,
                            f16* __restrict__ q16, f16* __restrict__ kT16,
                            f16* __restrict__ v16, int B, int Ksel, int H, int HD) {
  int half = HD >> 1;
  long tg = (long)blockIdx.x * blockDim.x + threadIdx.x;
  long total = (long)B * Ksel * H * half;
  if (tg >= total) return;
  int i = (int)(tg % half);
  long r = tg / half;
  int h = (int)(r % H);
  long bj = r / H;
  int j = (int)(bj % Ksel);
  int b = (int)(bj / Ksel);
  int pos = idx[(long)b * Ksel + j];
  float inv = powf(10000.f, -(2.f * i) / (float)HD);
  float ang = (float)pos * inv;
  float c = cosf(ang), s = sinf(ang);
  long rowin = bj * ((long)H * HD) + (long)h * HD;
  long bh = (long)b * H + h;
  long qo = (bh * Ksel + j) * HD;
  float x1 = (float)qp[rowin + i], x2 = (float)qp[rowin + i + half];
  q16[qo + i]        = (f16)(x1 * c - x2 * s);
  q16[qo + i + half] = (f16)(x2 * c + x1 * s);
  x1 = (float)kp[rowin + i]; x2 = (float)kp[rowin + i + half];
  long ko = bh * (long)HD * Ksel;
  kT16[ko + (long)i * Ksel + j]          = (f16)(x1 * c - x2 * s);
  kT16[ko + (long)(i + half) * Ksel + j] = (f16)(x2 * c + x1 * s);
  v16[qo + i]        = vp[rowin + i];
  v16[qo + i + half] = vp[rowin + i + half];
}

// Attention: one block = (b, h, 16-query tile). 4 waves. S tile in LDS.
__global__ void __launch_bounds__(128) attn_kernel(const f16* __restrict__ q16,
                                                   const f16* __restrict__ kT16,
                                                   const f16* __restrict__ v16,
                                                   f16* __restrict__ attn16,
                                                   int H, int S, int HD, int D) {
  int nqt = S >> 4;
  int qt = blockIdx.x % nqt;
  int bh = blockIdx.x / nqt;
  int b = bh / H, h = bh % H;
  int lane = threadIdx.x & 31, wave = threadIdx.x >> 5;
  __shared__ __align__(16) float Ssm[16 * 512];
  __shared__ __align__(16) f16   Psm[16 * 512];
  __shared__ float rmax[16 * 8], rsum[16 * 8];
  const f16* qb  = q16 + ((long)bh * S + qt * 16) * HD;
  const f16* kTb = kT16 + (long)bh * HD * S;
  const f16* vb  = v16 + (long)bh * S * HD;
  const float scale = 0.125f;   // 1/sqrt(64)

  for (int ntile = wave; ntile < (S >> 4); ntile += 4) {
    v8f acc = {};
    for (int kb = 0; kb < HD; kb += 32) {
      v16h a = load_a_frag(qb, HD, 0, kb, lane);
      v16h bfr = load_b_frag(kTb, S, kb, ntile * 16, lane);
      acc = wmma_f16(a, bfr, acc);
    }
    int r0 = (lane >> 4) << 3;
    int c = lane & 15;
    int keyj = ntile * 16 + c;
#pragma unroll
    for (int r = 0; r < 8; ++r) {
      int qrow = r0 + r;
      int qj = qt * 16 + qrow;
      float v = acc[r] * scale;
      if (keyj > qj) v = -1e9f;    // idx sorted strictly ascending -> causal == j_k <= j_q
      Ssm[qrow * 512 + keyj] = v;
    }
  }
  __syncthreads();

  int row = threadIdx.x >> 3, sub = threadIdx.x & 7;   // 8 threads x 64 cols per row
  float mx = -1e30f;
  for (int c0 = sub * 64; c0 < sub * 64 + 64; ++c0) mx = fmaxf(mx, Ssm[row * 512 + c0]);
  rmax[row * 8 + sub] = mx;
  __syncthreads();
  if (sub == 0) { float m = rmax[row * 8]; for (int i = 1; i < 8; ++i) m = fmaxf(m, rmax[row * 8 + i]); rmax[row * 8] = m; }
  __syncthreads();
  float m = rmax[row * 8];
  float sm = 0.f;
  for (int c0 = sub * 64; c0 < sub * 64 + 64; ++c0) {
    float e = expf(Ssm[row * 512 + c0] - m);
    Ssm[row * 512 + c0] = e; sm += e;
  }
  rsum[row * 8 + sub] = sm;
  __syncthreads();
  if (sub == 0) { float s = 0.f; for (int i = 0; i < 8; ++i) s += rsum[row * 8 + i]; rsum[row * 8] = s; }
  __syncthreads();
  float inv = 1.f / rsum[row * 8];
  for (int c0 = sub * 64; c0 < sub * 64 + 64; ++c0)
    Psm[row * 512 + c0] = (f16)(Ssm[row * 512 + c0] * inv);
  __syncthreads();

  {  // O = P @ V : HD/16 == 4 N-tiles == #waves
    int ntile = wave;
    v8f acc = {};
    for (int kb = 0; kb < S; kb += 32) {
      v16h a = load_a_frag((const f16*)Psm, 512, 0, kb, lane);
      v16h bfr = load_b_frag(vb, HD, kb, ntile * 16, lane);
      acc = wmma_f16(a, bfr, acc);
    }
    int r0 = (lane >> 4) << 3;
    int c = lane & 15;
#pragma unroll
    for (int r = 0; r < 8; ++r) {
      int qj = qt * 16 + r0 + r;
      attn16[((long)b * S + qj) * D + h * HD + ntile * 16 + c] = (f16)acc[r];
    }
  }
}

// h = sel + oproj; m16 = rmsnorm(h)*g2  (block per row)
__global__ void __launch_bounds__(256) resid_rms(const float* __restrict__ sel,
                                                 const float* __restrict__ oproj,
                                                 const float* __restrict__ g2,
                                                 float* __restrict__ hbuf,
                                                 f16* __restrict__ m16, int D) {
  long rowid = blockIdx.x;
  const float* s = sel + rowid * D;
  const float* o = oproj + rowid * D;
  float* hb = hbuf + rowid * D;
  f16* dm = m16 + rowid * D;
  int tid = threadIdx.x;
  float v[4]; float ss = 0.f;
#pragma unroll
  for (int c = 0; c < 4; ++c) { int i = tid + c * 256; float x = s[i] + o[i]; v[c] = x; hb[i] = x; ss += x * x; }
  __shared__ float red[8];
  __shared__ float stot;
  ss = wave_sum(ss);
  if ((tid & 31) == 0) red[tid >> 5] = ss;
  __syncthreads();
  if (tid == 0) { float t = 0.f; for (int i = 0; i < 8; ++i) t += red[i]; stot = t; }
  __syncthreads();
  float inv = rsqrtf(stot / (float)D + 1e-6f);
#pragma unroll
  for (int c = 0; c < 4; ++c) { int i = tid + c * 256; dm[i] = (f16)(v[c] * inv * g2[i]); }
}

__global__ void silu_mul(const f16* __restrict__ g, const f16* __restrict__ u,
                         f16* __restrict__ act, int n) {
  int i = blockIdx.x * blockDim.x + threadIdx.x;
  if (i >= n) return;
  float a = (float)g[i];
  act[i] = (f16)((a * sigmoidf_(a)) * (float)u[i]);
}

// out row = hidden row, or sel + gate*(h + down - sel) for selected rows
__global__ void __launch_bounds__(256) scatter_out(const float* __restrict__ hs,
                                                   const int* __restrict__ selpos,
                                                   const float* __restrict__ gate,
                                                   const float* __restrict__ hbuf,
                                                   const float* __restrict__ down,
                                                   float* __restrict__ out,
                                                   int T, int D, int Ksel) {
  long rowid = blockIdx.x;
  int b = (int)(rowid / T);
  int j = selpos[rowid];
  const float* src = hs + rowid * D;
  float* dst = out + rowid * D;
  int tid = threadIdx.x;
  if (j < 0) {
#pragma unroll
    for (int c = 0; c < 4; ++c) { int i = tid + c * 256; dst[i] = src[i]; }
  } else {
    long sr = ((long)b * Ksel + j) * D;
    float gt = gate[(long)b * Ksel + j];
#pragma unroll
    for (int c = 0; c < 4; ++c) {
      int i = tid + c * 256;
      float s = src[i];
      float bo = hbuf[sr + i] + down[sr + i];
      dst[i] = s + gt * (bo - s);
    }
  }
}

__global__ void finalize_loss(const float* __restrict__ accum, float* __restrict__ out_loss,
                              float invBT) {
  if (threadIdx.x == 0)
    out_loss[0] = accum[0] * invBT * 0.01f + accum[1] * invBT * 0.01f;
}

// ---------------------------------------------------------------------------
extern "C" void kernel_launch(void* const* d_in, const int* in_sizes, int n_in,
                              void* d_out, int out_size, void* d_ws, size_t ws_size,
                              hipStream_t stream) {
  const int B = 4, T = 4096, D = 1024, F = 4096, H = 16, HD = 64, S = 512, NP = 256;
  const int BT = B * T, BS = B * S;

  const float* hs   = (const float*)d_in[0];
  const float* rw   = (const float*)d_in[1];
  const float* fc1w = (const float*)d_in[2];
  const float* fc1b = (const float*)d_in[3];
  const float* fc2w = (const float*)d_in[4];
  const float* fc2b = (const float*)d_in[5];
  const float* ln1  = (const float*)d_in[6];
  const float* ln2  = (const float*)d_in[7];
  const float* wq   = (const float*)d_in[8];
  const float* wk   = (const float*)d_in[9];
  const float* wv   = (const float*)d_in[10];
  const float* wo   = (const float*)d_in[11];
  const float* wg   = (const float*)d_in[12];
  const float* wu   = (const float*)d_in[13];
  const float* wd   = (const float*)d_in[14];
  float* out = (float*)d_out;

  // ---- workspace carve (256B aligned) ----
  char* p = (char*)d_ws;
  auto carve = [&](size_t bytes) -> char* {
    char* r = p; p += (bytes + 255) & ~(size_t)255; return r;
  };
  f16* w16q   = (f16*)carve((size_t)D * D * 2);
  f16* w16k   = (f16*)carve((size_t)D * D * 2);
  f16* w16v   = (f16*)carve((size_t)D * D * 2);
  f16* w16o   = (f16*)carve((size_t)D * D * 2);
  f16* w16g   = (f16*)carve((size_t)D * F * 2);
  f16* w16u   = (f16*)carve((size_t)D * F * 2);
  f16* w16d   = (f16*)carve((size_t)F * D * 2);
  f16* w16fc1 = (f16*)carve((size_t)D * NP * 2);
  float* scores  = (float*)carve((size_t)BT * 4);
  float* targets = (float*)carve((size_t)BT * 4);
  int*   selpos  = (int*)carve((size_t)BT * 4);
  int*   idxbuf  = (int*)carve((size_t)BS * 4);
  float* gatebuf = (float*)carve((size_t)BS * 4);
  float* accum   = (float*)carve(256);
  f16* h16  = (f16*)carve((size_t)BT * D * 2);     // reused later for g16/u16
  f16* p1   = (f16*)carve((size_t)BT * NP * 2);
  float* sel = (float*)carve((size_t)BS * D * 4);
  f16* a16  = (f16*)carve((size_t)BS * D * 2);
  f16* qp16 = (f16*)carve((size_t)BS * D * 2);     // qp..v16 region reused for act16
  f16* kp16 = (f16*)carve((size_t)BS * D * 2);
  f16* vp16 = (f16*)carve((size_t)BS * D * 2);
  f16* q16  = (f16*)carve((size_t)B * H * S * HD * 2);
  f16* kT16 = (f16*)carve((size_t)B * H * S * HD * 2);
  f16* v16  = (f16*)carve((size_t)B * H * S * HD * 2);
  f16* attn16 = (f16*)carve((size_t)BS * D * 2);
  float* oproj = (float*)carve((size_t)BS * D * 4);  // reused for `down`
  float* hbuf  = (float*)carve((size_t)BS * D * 4);
  f16* m16  = (f16*)carve((size_t)BS * D * 2);
  // aliases (lifetimes verified by kernel ordering below)
  f16* g16   = h16;
  f16* u16   = h16 + (size_t)BS * F;
  f16* act16 = qp16;
  float* down = oproj;

  // ---- 1. weight / activation converts to f16 ----
  f32_to_f16<<<(D * D) / 256, 256, 0, stream>>>(wq, w16q, D * D);
  f32_to_f16<<<(D * D) / 256, 256, 0, stream>>>(wk, w16k, D * D);
  f32_to_f16<<<(D * D) / 256, 256, 0, stream>>>(wv, w16v, D * D);
  f32_to_f16<<<(D * D) / 256, 256, 0, stream>>>(wo, w16o, D * D);
  f32_to_f16<<<(D * F) / 256, 256, 0, stream>>>(wg, w16g, D * F);
  f32_to_f16<<<(D * F) / 256, 256, 0, stream>>>(wu, w16u, D * F);
  f32_to_f16<<<(F * D) / 256, 256, 0, stream>>>(wd, w16d, F * D);
  f32_to_f16<<<(D * NP) / 256, 256, 0, stream>>>(fc1w, w16fc1, D * NP);
  f32_to_f16<<<(BT * D) / 256, 256, 0, stream>>>(hs, h16, BT * D);
  init_accum<<<1, 32, 0, stream>>>(accum);

  // ---- 2. router + top-k ----
  router_scores<<<BT / 8, 256, 0, stream>>>(hs, rw, scores, BT, D);
  topk_kernel<<<B, 1024, 0, stream>>>(scores, idxbuf, gatebuf, selpos, targets, T, S);

  // ---- 3. predictor GEMM + losses (before h16 is reused) ----
  gemm_lds<<<(BT / 64) * (NP / 128), 256, 0, stream>>>(h16, w16fc1, nullptr, p1, BT, NP, D);
  loss_kernel<<<BT / 8, 256, 0, stream>>>(p1, fc1b, fc2w, fc2b, scores, targets, accum, BT, NP);

  // ---- 4. gather + RMSNorm1 ----
  gather_rms<<<BS, 256, 0, stream>>>(hs, idxbuf, ln1, sel, a16, T, D, S);

  // ---- 5. QKV projections (WMMA, LDS-tiled) ----
  int qkvBlocks = (BS / 64) * (D / 128);
  gemm_lds<<<qkvBlocks, 256, 0, stream>>>(a16, w16q, nullptr, qp16, BS, D, D);
  gemm_lds<<<qkvBlocks, 256, 0, stream>>>(a16, w16k, nullptr, kp16, BS, D, D);
  gemm_lds<<<qkvBlocks, 256, 0, stream>>>(a16, w16v, nullptr, vp16, BS, D, D);

  // ---- 6. RoPE + relayout ----
  rope_kernel<<<(B * S * H * (HD / 2)) / 256, 256, 0, stream>>>(
      qp16, kp16, vp16, idxbuf, q16, kT16, v16, B, S, H, HD);

  // ---- 7. attention (WMMA QK^T + WMMA PV) ----
  attn_kernel<<<B * H * (S / 16), 128, 0, stream>>>(q16, kT16, v16, attn16, H, S, HD, D);

  // ---- 8. output projection + residual + RMSNorm2 ----
  gemm_lds<<<qkvBlocks, 256, 0, stream>>>(attn16, w16o, oproj, nullptr, BS, D, D);
  resid_rms<<<BS, 256, 0, stream>>>(sel, oproj, ln2, hbuf, m16, D);

  // ---- 9. MLP (WMMA, LDS-tiled) ----
  int guBlocks = (BS / 64) * (F / 128);
  gemm_lds<<<guBlocks, 256, 0, stream>>>(m16, w16g, nullptr, g16, BS, F, D);
  gemm_lds<<<guBlocks, 256, 0, stream>>>(m16, w16u, nullptr, u16, BS, F, D);
  silu_mul<<<(BS * F) / 256, 256, 0, stream>>>(g16, u16, act16, BS * F);
  gemm_lds<<<(BS / 64) * (D / 128), 256, 0, stream>>>(act16, w16d, down, nullptr, BS, D, F);

  // ---- 10. scatter + loss ----
  scatter_out<<<BT, 256, 0, stream>>>(hs, selpos, gatebuf, hbuf, down, out, T, D, S);
  finalize_loss<<<1, 32, 0, stream>>>(accum, out + (size_t)BT * D, 1.f / (float)BT);

  (void)in_sizes; (void)n_in; (void)out_size; (void)ws_size;
}
